// TensorOutputLayer_26757646254492
// MI455X (gfx1250) — matrix operand order repaired and checked by reading, not compiled, not run on hardware
//
#include <hip/hip_runtime.h>

// ---------------------------------------------------------------------------
// out[4096,1000] = core[4096, K=4096] @ weights[1000, K]^T   (fp32 in/out)
//
// MI455X plan:
//  * Compute-bound GEMM (AI ~350 FLOP/B) -> v_wmma_f32_16x16x32_bf16.
//  * fp32 emulated as bf16 hi+lo (3 WMMAs per K=32: hi*hi + lo*hi + hi*lo),
//    ~2^-15 relative error, ~2.7x the throughput of exact 16x16x4 f32 WMMA.
//  * Pass 1 (memory-bound, ~7us): split fp32 -> packed bf16 hi/lo planes in
//    d_ws so the GEMM inner loop carries ZERO conversion VALU.
//  * Pass 2: 8 waves/WG (2x4), wave = 4x2 tiles of 16x16 (64x32), block
//    128x128, grid 8x32 = 2048 waves (2/SIMD). bf16 planes (84MB) stay
//    L2-resident (192MB L2), so cross-block re-reads never touch HBM.
//  * Addressing: uniform SGPR base pointers advanced 32 elems/K-step (SALU),
//    constant 32-bit per-lane voffsets -> saddr+voffset+ioffset loads, no
//    64-bit VGPR address adds (and none of their WMMA-hazard v_nops).
//  * Fallback (ws too small): fused convert-in-loop GEMM.
// ---------------------------------------------------------------------------

typedef __attribute__((ext_vector_type(16))) __bf16    v16bf;
typedef __attribute__((ext_vector_type(8)))  __bf16    v8bf;
typedef __attribute__((ext_vector_type(8)))  float     v8f;
typedef __attribute__((ext_vector_type(8)))  unsigned  v8u;

#define MDIM 4096
#define NDIM 1000
#define KDIM 4096

__device__ __forceinline__ unsigned pk_hi_bf16(float a, float b) {
  unsigned ua = __builtin_bit_cast(unsigned, a);
  unsigned ub = __builtin_bit_cast(unsigned, b);
  return (ua >> 16) | (ub & 0xFFFF0000u);
}
__device__ __forceinline__ float trunc_bf16(float a) {
  unsigned ua = __builtin_bit_cast(unsigned, a) & 0xFFFF0000u;
  return __builtin_bit_cast(float, ua);
}
__device__ __forceinline__ v16bf cat8(v8bf a, v8bf b) {
  return __builtin_shufflevector(a, b, 0, 1, 2, 3, 4, 5, 6, 7,
                                 8, 9, 10, 11, 12, 13, 14, 15);
}

// ------------------------- Pass 1: fp32 -> bf16 hi/lo ----------------------
__global__ __launch_bounds__(256) void
convert_split_bf16(const float* __restrict__ src, __bf16* __restrict__ dhi,
                   __bf16* __restrict__ dlo, int n8) {
  int t = blockIdx.x * blockDim.x + threadIdx.x;
  if (t >= n8) return;
  size_t idx = (size_t)t * 8;
  const float4* p = (const float4*)(src + idx);
  float4 f0 = p[0], f1 = p[1];
  float f[8] = {f0.x, f0.y, f0.z, f0.w, f1.x, f1.y, f1.z, f1.w};
  unsigned hv[4], lv[4];
#pragma unroll
  for (int q = 0; q < 4; ++q) {
    float a = f[2 * q], b = f[2 * q + 1];
    hv[q] = pk_hi_bf16(a, b);
    lv[q] = pk_hi_bf16(a - trunc_bf16(a), b - trunc_bf16(b));
  }
  *(uint4*)(dhi + idx) = make_uint4(hv[0], hv[1], hv[2], hv[3]);
  *(uint4*)(dlo + idx) = make_uint4(lv[0], lv[1], lv[2], lv[3]);
}

// ------------------------- Pass 2: pure-WMMA GEMM --------------------------
__global__ __launch_bounds__(256) void
gemm_bf16split_pre(const __bf16* __restrict__ aHi, const __bf16* __restrict__ aLo,
                   const __bf16* __restrict__ bHi, const __bf16* __restrict__ bLo,
                   float* __restrict__ out) {
  const int lane     = threadIdx.x & 31;
  const int wave     = threadIdx.x >> 5;
  const int lane16   = lane & 15;
  const int laneHalf = lane >> 4;

  const int wave_m = wave & 1;
  const int wave_n = wave >> 1;
  const int m_w = blockIdx.y * 128 + wave_m * 64;
  const int n_w = blockIdx.x * 128 + wave_n * 32;

  // 32-bit per-lane element indices (byte offsets < 2^26, well within GVS
  // signed-32 voffset range). Chunk offsets (+8 K / +16 K) fold into ioffset.
  unsigned aIdx[4];
#pragma unroll
  for (int i = 0; i < 4; ++i)
    aIdx[i] = (unsigned)(m_w + 16 * i + lane16) * KDIM + (unsigned)laneHalf * 8;
  unsigned bIdx[2];
#pragma unroll
  for (int j = 0; j < 2; ++j) {
    int col = n_w + 16 * j + lane16;
    col = col < NDIM ? col : NDIM - 1;           // clamp padded classes
    bIdx[j] = (unsigned)col * KDIM + (unsigned)laneHalf * 8;
  }

  // Uniform bases advanced by SALU each K-step.
  const __bf16* pAh = aHi;
  const __bf16* pAl = aLo;
  const __bf16* pBh = bHi;
  const __bf16* pBl = bLo;

  v8f acc[4][2] = {};

  for (int k0 = 0; k0 < KDIM; k0 += 32) {
    v16bf ahi[4], alo[4], bhi[2], blo[2];

    // A 16x32 bf16 layout: lanes 0-15 take K[k0..+7] & K[k0+16..+23];
    // lanes 16-31 take K[k0+8..+15] & K[k0+24..+31]. 8 bf16 = one b128.
#pragma unroll
    for (int i = 0; i < 4; ++i) {
      const v8bf* ph = (const v8bf*)(pAh + aIdx[i]);
      const v8bf* pl = (const v8bf*)(pAl + aIdx[i]);
      ahi[i] = cat8(ph[0], ph[2]);   // ioffset 0 and +32B (16 elems)
      alo[i] = cat8(pl[0], pl[2]);
    }
    // B 32x16 layout: lane16 = column, same K chunks as A per lane-half.
#pragma unroll
    for (int j = 0; j < 2; ++j) {
      const v8bf* ph = (const v8bf*)(pBh + bIdx[j]);
      const v8bf* pl = (const v8bf*)(pBl + bIdx[j]);
      bhi[j] = cat8(ph[0], ph[2]);   // ioffset 0 and +32B (16 elems)
      blo[j] = cat8(pl[0], pl[2]);
    }

    // 3 emulation terms interleaved across 8 independent accumulators.
#pragma unroll
    for (int i = 0; i < 4; ++i)
#pragma unroll
      for (int j = 0; j < 2; ++j)
        acc[i][j] = __builtin_amdgcn_wmma_f32_16x16x32_bf16(
            false, ahi[i], false, bhi[j], (short)0, acc[i][j], false, false);
#pragma unroll
    for (int i = 0; i < 4; ++i)
#pragma unroll
      for (int j = 0; j < 2; ++j)
        acc[i][j] = __builtin_amdgcn_wmma_f32_16x16x32_bf16(
            false, alo[i], false, bhi[j], (short)0, acc[i][j], false, false);
#pragma unroll
    for (int i = 0; i < 4; ++i)
#pragma unroll
      for (int j = 0; j < 2; ++j)
        acc[i][j] = __builtin_amdgcn_wmma_f32_16x16x32_bf16(
            false, ahi[i], false, blo[j], (short)0, acc[i][j], false, false);

    pAh += 32; pAl += 32; pBh += 32; pBl += 32;
  }

  // C/D layout: VGPR r -> row (r + 8*laneHalf), col = lane16.
#pragma unroll
  for (int i = 0; i < 4; ++i)
#pragma unroll
    for (int j = 0; j < 2; ++j) {
      int col = n_w + 16 * j + lane16;
      if (col < NDIM) {
#pragma unroll
        for (int r = 0; r < 8; ++r) {
          int row = m_w + 16 * i + r + laneHalf * 8;
          out[(size_t)row * NDIM + col] = acc[i][j][r];
        }
      }
    }
}

// --------------- Fallback: fused convert-in-loop GEMM (no ws) --------------
__device__ __forceinline__ void build_frag(float4 c0, float4 c1, float4 c2,
                                           float4 c3, v16bf& hi, v16bf& lo) {
  float f[16] = {c0.x, c0.y, c0.z, c0.w, c1.x, c1.y, c1.z, c1.w,
                 c2.x, c2.y, c2.z, c2.w, c3.x, c3.y, c3.z, c3.w};
  v8u h, l;
#pragma unroll
  for (int t = 0; t < 8; ++t) {
    float a = f[2 * t], b = f[2 * t + 1];
    h[t] = pk_hi_bf16(a, b);
    l[t] = pk_hi_bf16(a - trunc_bf16(a), b - trunc_bf16(b));
  }
  hi = __builtin_bit_cast(v16bf, h);
  lo = __builtin_bit_cast(v16bf, l);
}

__global__ __launch_bounds__(256) void
gemm_bf16split_direct(const float* __restrict__ core,
                      const float* __restrict__ weights,
                      float* __restrict__ out) {
  const int lane     = threadIdx.x & 31;
  const int wave     = threadIdx.x >> 5;
  const int lane16   = lane & 15;
  const int laneHalf = lane >> 4;
  const int wave_m = wave & 1;
  const int wave_n = wave >> 1;
  const int m_w = blockIdx.y * 128 + wave_m * 64;
  const int n_w = blockIdx.x * 128 + wave_n * 32;

  unsigned aIdx[4];
#pragma unroll
  for (int i = 0; i < 4; ++i)
    aIdx[i] = (unsigned)(m_w + 16 * i + lane16) * KDIM + (unsigned)laneHalf * 8;
  unsigned bIdx[2];
#pragma unroll
  for (int j = 0; j < 2; ++j) {
    int col = n_w + 16 * j + lane16;
    col = col < NDIM ? col : NDIM - 1;
    bIdx[j] = (unsigned)col * KDIM + (unsigned)laneHalf * 8;
  }
  const float* pA = core;
  const float* pB = weights;

  v8f acc[4][2] = {};
  for (int k0 = 0; k0 < KDIM; k0 += 32) {
    v16bf ahi[4], alo[4], bhi[2], blo[2];
#pragma unroll
    for (int i = 0; i < 4; ++i) {
      const float4* p = (const float4*)(pA + aIdx[i]);
      build_frag(p[0], p[1], p[4], p[5], ahi[i], alo[i]);  // K +0..7, +16..23
    }
#pragma unroll
    for (int j = 0; j < 2; ++j) {
      const float4* p = (const float4*)(pB + bIdx[j]);
      build_frag(p[0], p[1], p[4], p[5], bhi[j], blo[j]);  // K +0..7, +16..23
    }
#pragma unroll
    for (int i = 0; i < 4; ++i)
#pragma unroll
      for (int j = 0; j < 2; ++j)
        acc[i][j] = __builtin_amdgcn_wmma_f32_16x16x32_bf16(
            false, ahi[i], false, bhi[j], (short)0, acc[i][j], false, false);
#pragma unroll
    for (int i = 0; i < 4; ++i)
#pragma unroll
      for (int j = 0; j < 2; ++j)
        acc[i][j] = __builtin_amdgcn_wmma_f32_16x16x32_bf16(
            false, alo[i], false, bhi[j], (short)0, acc[i][j], false, false);
#pragma unroll
    for (int i = 0; i < 4; ++i)
#pragma unroll
      for (int j = 0; j < 2; ++j)
        acc[i][j] = __builtin_amdgcn_wmma_f32_16x16x32_bf16(
            false, ahi[i], false, blo[j], (short)0, acc[i][j], false, false);
    pA += 32; pB += 32;
  }

#pragma unroll
  for (int i = 0; i < 4; ++i)
#pragma unroll
    for (int j = 0; j < 2; ++j) {
      int col = n_w + 16 * j + lane16;
      if (col < NDIM) {
#pragma unroll
        for (int r = 0; r < 8; ++r) {
          int row = m_w + 16 * i + r + laneHalf * 8;
          out[(size_t)row * NDIM + col] = acc[i][j][r];
        }
      }
    }
}

// ---------------------------------------------------------------------------
extern "C" void kernel_launch(void* const* d_in, const int* in_sizes, int n_in,
                              void* d_out, int out_size, void* d_ws, size_t ws_size,
                              hipStream_t stream) {
  (void)in_sizes; (void)n_in; (void)out_size;
  const float* core    = (const float*)d_in[0];   // [4096, 16,16,16] fp32
  const float* weights = (const float*)d_in[1];   // [1000, 16,16,16] fp32
  float* out = (float*)d_out;                     // [4096, 1000] fp32

  const size_t aElems = (size_t)MDIM * KDIM;      // 16,777,216
  const size_t bElems = (size_t)NDIM * KDIM;      //  4,096,000
  const size_t need   = 2 * 2 * (aElems + bElems); // hi+lo bf16 = 83.5 MB

  dim3 grid((NDIM + 127) / 128, MDIM / 128);      // (8, 32) blocks of 128x128

  if (ws_size >= need) {
    __bf16* aHi = (__bf16*)d_ws;
    __bf16* aLo = aHi + aElems;
    __bf16* bHi = aLo + aElems;
    __bf16* bLo = bHi + bElems;

    int n8a = (int)(aElems / 8);
    int n8b = (int)(bElems / 8);
    convert_split_bf16<<<(n8a + 255) / 256, 256, 0, stream>>>(core, aHi, aLo, n8a);
    convert_split_bf16<<<(n8b + 255) / 256, 256, 0, stream>>>(weights, bHi, bLo, n8b);
    gemm_bf16split_pre<<<grid, dim3(256), 0, stream>>>(aHi, aLo, bHi, bLo, out);
  } else {
    gemm_bf16split_direct<<<grid, dim3(256), 0, stream>>>(core, weights, out);
  }
}
